// Model_18030272708988
// MI455X (gfx1250) — compile-verified, matrix operand order checked by reading, not run on hardware
//
#include <hip/hip_runtime.h>

typedef __attribute__((ext_vector_type(16))) _Float16 v16h;
typedef __attribute__((ext_vector_type(8)))  float    v8f;
typedef __attribute__((ext_vector_type(4)))  float    v4f;

#define N_SAMPLE 8192
#define N_OBS    4096

#define LN2F      0.6931471805599453f
#define INV_LN2F  1.4426950408889634f
#define HL2P      0.9189385332046727f

// workspace layout (float offsets)
#define OFF_SLOGY 0                       // 1 float: sum of ln(y)
#define OFF_L2Y   16                      // 4096 floats: log2(y_t)
#define OFF_LBD   (OFF_L2Y + N_OBS)       // 8192 floats: lbd_s
#define OFF_CS    (OFF_LBD + N_SAMPLE)    // 8192 floats: k^{-lbd}/S
#define OFF_PEN   (OFF_CS + N_SAMPLE)     // 32 floats: per-block (pen-base)/S partials
#define OFF_MAIN  (OFF_PEN + 32)          // 64 floats: per-WG Σ cs*Ssum partials

__device__ __forceinline__ float fast_sigmoid(float x) {
    // 1 / (1 + exp(-x)) with hardware exp2 + rcp
    return __builtin_amdgcn_rcpf(1.0f + __builtin_amdgcn_exp2f(-x * INV_LN2F));
}

// ---------------------------------------------------------------------------
// Kernel 1: l2y[t] = log2(y[t]); ws[0] = Σ ln(y_t).  One block, 1024 threads.
// ---------------------------------------------------------------------------
__global__ __launch_bounds__(1024) void weibull_prep(const float* __restrict__ y,
                                                     float* __restrict__ ws) {
    __shared__ float sm[1024];
    const int tid = threadIdx.x;
    float acc = 0.0f;
#pragma unroll
    for (int j = 0; j < 4; ++j) {
        const int t = tid + j * 1024;
        const float l2 = __builtin_amdgcn_logf(y[t]);   // v_log_f32 == log2
        ws[OFF_L2Y + t] = l2;
        acc += l2;
    }
    sm[tid] = acc;
    __syncthreads();
    for (int off = 512; off > 0; off >>= 1) {
        if (tid < off) sm[tid] += sm[tid + off];
        __syncthreads();
    }
    if (tid == 0) ws[OFF_SLOGY] = sm[0] * LN2F;         // Σ ln y = ln2 * Σ log2 y
}

// ---------------------------------------------------------------------------
// Kernel 2: per-sample lbd, cs = k^{-lbd}/S, and (penalty - base)/S partials.
// 32 blocks x 256 threads.
// ---------------------------------------------------------------------------
__global__ __launch_bounds__(256) void weibull_sparams(
    const float* __restrict__ param,
    const float* __restrict__ ea, const float* __restrict__ e0,
    const float* __restrict__ ei, const float* __restrict__ ed,
    const float* __restrict__ iv, const float* __restrict__ dv,
    float* __restrict__ ws) {
    const int s = blockIdx.x * 256 + threadIdx.x;

    const float a_mu = param[0], a_lv = param[1];
    const float b0_mu = param[2], b0_lv = param[3];
    const float bi_mu = param[4], bi_lv = param[5];
    const float bd_mu = param[6], bd_lv = param[7];
    const float i_ = iv[0], d_ = dv[0];
    const float SlogY = ws[OFF_SLOGY];

    const float a_sd  = __builtin_amdgcn_exp2f(0.5f * a_lv  * INV_LN2F);
    const float b0_sd = __builtin_amdgcn_exp2f(0.5f * b0_lv * INV_LN2F);
    const float bi_sd = __builtin_amdgcn_exp2f(0.5f * bi_lv * INV_LN2F);
    const float bd_sd = __builtin_amdgcn_exp2f(0.5f * bd_lv * INV_LN2F);

    const float epsa = ea[s], eps0 = e0[s], epsi = ei[s], epsd = ed[s];
    const float alpha  = epsa * a_sd  + a_mu;
    const float beta0  = eps0 * b0_sd + b0_mu;
    const float betai  = epsi * bi_sd + bi_mu;
    const float betad  = epsd * bd_sd + bd_mu;

    const float k   = 2.0f * fast_sigmoid(alpha);
    const float lbd = 2.0f * fast_sigmoid(-(beta0 + betai * i_ + betad * d_));

    const float l2k = __builtin_amdgcn_logf(k);
    const float l2l = __builtin_amdgcn_logf(lbd);
    const float T = (float)N_OBS;

    // Σ_t ll = T ln(lbd) - T lbd ln(k) + (lbd-1) SlogY - k^{-lbd} * Ssum
    const float base = T * LN2F * l2l - T * lbd * LN2F * l2k + (lbd - 1.0f) * SlogY;

    // penalty with (x-mu)/sd == eps exactly: -0.5*lv - HL2P - 0.5*eps^2, x4
    const float pen = -0.5f * (a_lv + b0_lv + bi_lv + bd_lv) - 4.0f * HL2P
                      - 0.5f * (epsa * epsa + eps0 * eps0 + epsi * epsi + epsd * epsd);

    const float Cs = __builtin_amdgcn_exp2f(-lbd * l2k);  // k^{-lbd}

    ws[OFF_LBD + s] = lbd;
    ws[OFF_CS  + s] = Cs * (1.0f / (float)N_SAMPLE);

    // block-reduce (pen - base)/S
    __shared__ float sm[256];
    sm[threadIdx.x] = (pen - base) * (1.0f / (float)N_SAMPLE);
    __syncthreads();
    for (int off = 128; off > 0; off >>= 1) {
        if (threadIdx.x < off) sm[threadIdx.x] += sm[threadIdx.x + off];
        __syncthreads();
    }
    if (threadIdx.x == 0) ws[OFF_PEN + blockIdx.x] = sm[0];
}

// ---------------------------------------------------------------------------
// Kernel 3: Ssum_s = Σ_t exp2(lbd_s * log2 y_t), reduced with WMMA (B = ones).
// 64 WGs x 256 threads; each wave owns 16 samples; K=32 observations / WMMA.
// A-matrix 16x32 f16 layout: lane (m = lane&15, g = lane>>4) holds
//   K in {8g..8g+7} and {16+8g..16+8g+7}.
// ---------------------------------------------------------------------------
__global__ __launch_bounds__(256) void weibull_main(const float* __restrict__ ws,
                                                    float* __restrict__ wg_part) {
    __shared__ __align__(16) float l2y[N_OBS];   // 16 KB staged table
    __shared__ float red[16];

    const int tid = threadIdx.x;

    // stage log2(y) into LDS: 1024 float4s / 256 threads
    const v4f* src = (const v4f*)(ws + OFF_L2Y);
    v4f* dst = (v4f*)l2y;
#pragma unroll
    for (int j = 0; j < 4; ++j) {
        const int idx = tid + j * 256;
        dst[idx] = src[idx];
    }
    __syncthreads();

    const int lane = tid & 31;
    const int wave = tid >> 5;
    const int sb = (blockIdx.x * 8 + wave) * 16;   // 16-sample block for this wave
    const int m = lane & 15;
    const int g = lane >> 4;

    const float lbdm = ws[OFF_LBD + sb + m];

    v16h bones;
#pragma unroll
    for (int j = 0; j < 16; ++j) bones[j] = (_Float16)1.0f;

    v8f c = {};  // f32 accumulators: D[m][n] identical across n == Ssum

    const float* plo = &l2y[8 * g];
    const float* phi = &l2y[16 + 8 * g];

    for (int t = 0; t < N_OBS; t += 32) {
        const v4f lo0 = *(const v4f*)(plo + t);
        const v4f lo1 = *(const v4f*)(plo + t + 4);
        const v4f hi0 = *(const v4f*)(phi + t);
        const v4f hi1 = *(const v4f*)(phi + t + 4);
        v16h a;
#pragma unroll
        for (int j = 0; j < 4; ++j) {
            a[j]      = (_Float16)__builtin_amdgcn_exp2f(lbdm * lo0[j]);
            a[4 + j]  = (_Float16)__builtin_amdgcn_exp2f(lbdm * lo1[j]);
            a[8 + j]  = (_Float16)__builtin_amdgcn_exp2f(lbdm * hi0[j]);
            a[12 + j] = (_Float16)__builtin_amdgcn_exp2f(lbdm * hi1[j]);
        }
        // D = A x ones + C : every column of D row m accumulates Σ_k exp2(...)
        c = __builtin_amdgcn_wmma_f32_16x16x32_f16(
            /*neg_a=*/false, a, /*neg_b=*/false, bones,
            /*c_mod=*/(short)0, c, /*reuse_a=*/false, /*reuse_b=*/false);
    }

    // C/D layout: VGPR r = rows (r, r+8); lanes 0-15 -> M=r, lanes 16-31 -> M=r+8
    if (lane == 0) {
        float part = 0.0f;
#pragma unroll
        for (int r = 0; r < 8; ++r) part += ws[OFF_CS + sb + r] * c[r];
        red[wave * 2] = part;
    }
    if (lane == 16) {
        float part = 0.0f;
#pragma unroll
        for (int r = 0; r < 8; ++r) part += ws[OFF_CS + sb + 8 + r] * c[r];
        red[wave * 2 + 1] = part;
    }
    __syncthreads();
    if (tid == 0) {
        float s = 0.0f;
#pragma unroll
        for (int j = 0; j < 16; ++j) s += red[j];
        wg_part[blockIdx.x] = s;
    }
}

// ---------------------------------------------------------------------------
// Kernel 4: deterministic final sum of 32 + 64 partials.
// ---------------------------------------------------------------------------
__global__ void weibull_final(const float* __restrict__ ws, float* __restrict__ out) {
    if (threadIdx.x == 0) {
        float s = 0.0f;
        for (int j = 0; j < 32; ++j) s += ws[OFF_PEN + j];
        for (int j = 0; j < 64; ++j) s += ws[OFF_MAIN + j];
        out[0] = s;
    }
}

extern "C" void kernel_launch(void* const* d_in, const int* in_sizes, int n_in,
                              void* d_out, int out_size, void* d_ws, size_t ws_size,
                              hipStream_t stream) {
    (void)in_sizes; (void)n_in; (void)out_size; (void)ws_size;
    const float* param = (const float*)d_in[0];
    const float* ea    = (const float*)d_in[1];
    const float* e0    = (const float*)d_in[2];
    const float* ei    = (const float*)d_in[3];
    const float* ed    = (const float*)d_in[4];
    const float* iv    = (const float*)d_in[5];
    const float* dv    = (const float*)d_in[6];
    const float* y     = (const float*)d_in[7];
    float* ws  = (float*)d_ws;
    float* out = (float*)d_out;

    weibull_prep<<<1, 1024, 0, stream>>>(y, ws);
    weibull_sparams<<<N_SAMPLE / 256, 256, 0, stream>>>(param, ea, e0, ei, ed, iv, dv, ws);
    weibull_main<<<N_SAMPLE / 128, 256, 0, stream>>>(ws, ws + OFF_MAIN);
    weibull_final<<<1, 32, 0, stream>>>(ws, out);
}
